// SLDM_85658827751702
// MI455X (gfx1250) — matrix operand order repaired and checked by reading, not compiled, not run on hardware
//
#include <hip/hip_runtime.h>

// ---------------------------------------------------------------------------
// Problem constants (match reference)
// ---------------------------------------------------------------------------
#define KB   8
#define BB   64
#define TT   128
#define YY   128
#define XX   64
#define HYH  256
#define HXH  256
#define BT   (BB * TT)          // 8192

typedef __bf16 bf16;
typedef __bf16 bf16x8  __attribute__((ext_vector_type(8)));
typedef __bf16 bf16x16 __attribute__((ext_vector_type(16)));
typedef float  v8f     __attribute__((ext_vector_type(8)));

// ---------------------------------------------------------------------------
// WMMA helpers (CDNA5 wave32, v_wmma_f32_16x16x32_bf16)
// A-matrix 16x32 bf16: lane = 16*hi + lr holds row M=lr;
//   frag elems 0..7  -> K = k0 + 8*hi + e
//   frag elems 8..15 -> K = k0 + 8*hi + 16 + (e-8)
// B-matrix 32x16 bf16 ("row striped across lanes"): lane holds col N=lr,
//   frag elems e -> same K mapping as A (K = k0 + 8*hi + e, +16 for e>=8)
// C/D 16x16 f32: elem r -> M = r + 8*hi, N = lr
// ---------------------------------------------------------------------------
__device__ __forceinline__ bf16x16 load_a_frag(const bf16* base, int ld,
                                               int row0, int k0, int lane) {
  int lr = lane & 15, hi = lane >> 4;
  const bf16* p = base + (long)(row0 + lr) * ld + k0 + hi * 8;
  bf16x8 lo = *(const bf16x8*)p;
  bf16x8 hb = *(const bf16x8*)(p + 16);
  bf16x16 f;
#pragma unroll
  for (int i = 0; i < 8; ++i) { f[i] = lo[i]; f[i + 8] = hb[i]; }
  return f;
}

__device__ __forceinline__ bf16x16 load_b_frag(const bf16* base, int ld,
                                               int row0, int k0, int lane) {
  int lr = lane & 15, hi = lane >> 4;
  const bf16* p = base + (long)(row0 + lr) * ld + k0 + hi * 8;
  bf16x8 lo = *(const bf16x8*)p;
  bf16x8 hb = *(const bf16x8*)(p + 16);
  bf16x16 f;
#pragma unroll
  for (int i = 0; i < 8; ++i) { f[i] = lo[i]; f[i + 8] = hb[i]; }
  return f;
}

__device__ __forceinline__ v8f wmma_bf16(bf16x16 a, bf16x16 b, v8f c) {
  return __builtin_amdgcn_wmma_f32_16x16x32_bf16(
      /*neg_a=*/false, a, /*neg_b=*/false, b,
      /*c_mod=*/(short)0, c, /*reuse_a=*/false, /*reuse_b=*/false);
}

// ---------------------------------------------------------------------------
// Small elementwise kernels
// ---------------------------------------------------------------------------
__global__ void f32_to_bf16_kernel(const float* __restrict__ src,
                                   bf16* __restrict__ dst, long n) {
  long i = (long)blockIdx.x * blockDim.x + threadIdx.x;
  if (i < n) dst[i] = (bf16)src[i];
}

__global__ void add_vec_kernel(const float* __restrict__ a,
                               const float* __restrict__ b,
                               float* __restrict__ c, int n) {
  int i = blockIdx.x * blockDim.x + threadIdx.x;
  if (i < n) c[i] = a[i] + b[i];
}

__global__ void fill_zero_kernel(float* __restrict__ p, long n) {
  long i = (long)blockIdx.x * blockDim.x + threadIdx.x;
  if (i < n) p[i] = 0.0f;
}

// ---------------------------------------------------------------------------
// Generic batched GEMM:  C[bz] = A[bz] @ B[bz]^T (+ bias) (+relu)
//   A: [M, Kd] bf16 row-major (lda), B: [N, Kd] bf16 row-major (ldb)
//   C: [M, N] f32 (ldc).  flags bit0 = relu.
//   flags bit1: also write aux = bf16(C + eps) with identical geometry to C.
// Block: 256 thr (8 waves) -> 64x64 C tile; wave computes 1 Mx16 row pair
// (shares A-fragment across two N tiles).  Grid: (M/64, N/64, batch).
// ---------------------------------------------------------------------------
__global__ __launch_bounds__(256) void gemm_bt_kernel(
    const bf16* __restrict__ A, const bf16* __restrict__ Bm,
    const float* __restrict__ bias, float* __restrict__ C,
    bf16* __restrict__ aux, const float* __restrict__ eps,
    int Kd, int lda, int ldb, int ldc,
    long sA, long sB, long sBias, long sC, int flags) {
  long bz = blockIdx.z;
  A  += bz * sA;
  Bm += bz * sB;
  C  += bz * sC;
  const float* biasz = bias ? bias + bz * sBias : nullptr;
  const float* epsz  = (flags & 2) ? eps + bz * sC : nullptr;
  bf16*        auxz  = (flags & 2) ? aux + bz * sC : nullptr;

  int tid = threadIdx.x, w = tid >> 5, lane = tid & 31;
  int m0    = blockIdx.x * 64 + (w >> 1) * 16;
  int nbase = blockIdx.y * 64 + (w & 1) * 32;

  v8f acc0 = {0.f, 0.f, 0.f, 0.f, 0.f, 0.f, 0.f, 0.f};
  v8f acc1 = acc0;
  for (int k0 = 0; k0 < Kd; k0 += 32) {
    bf16x16 a  = load_a_frag(A, lda, m0, k0, lane);
    bf16x16 b0 = load_b_frag(Bm, ldb, nbase, k0, lane);
    bf16x16 b1 = load_b_frag(Bm, ldb, nbase + 16, k0, lane);
    acc0 = wmma_bf16(a, b0, acc0);
    acc1 = wmma_bf16(a, b1, acc1);
  }
  int lr = lane & 15, hi = lane >> 4;
#pragma unroll
  for (int tt = 0; tt < 2; ++tt) {
    v8f acc = tt ? acc1 : acc0;
    int n = nbase + tt * 16 + lr;
    float bv = biasz ? biasz[n] : 0.0f;
#pragma unroll
    for (int r = 0; r < 8; ++r) {
      int m = m0 + r + hi * 8;
      long off = (long)m * ldc + n;
      float v = acc[r] + bv;
      if (flags & 1) v = fmaxf(v, 0.0f);
      C[off] = v;
      if (flags & 2) auxz[off] = (bf16)(v + epsz[off]);
    }
  }
}

// ---------------------------------------------------------------------------
// Backward ReLU-RNN recurrence, one (state k, batch-chunk of 16) per block.
//   h[t] = relu(xW[t] + h[t+1] @ Whh^T),  h[T] = h0   (biases folded in xW)
// Whh (256x256 bf16, 128KB) staged in LDS; h state double-buffered in LDS.
// grid = (4 batch chunks, K).  Block 256 thr = 8 waves; wave owns 2 N-tiles.
// ---------------------------------------------------------------------------
#define HREC 256
#define LDP  264           // padded LDS row (halves): 528B stride, bank-friendly
#define RECROWS 16

__global__ __launch_bounds__(256) void rec_kernel(
    const bf16* __restrict__ Whh, long whh_kstride,
    const float* __restrict__ xW, long xw_kstride, int ldx,
    const float* __restrict__ h0,           // [K][HREC]
    bf16* __restrict__ hout, long ho_kstride, int ldo, int T) {
  extern __shared__ char smem_raw[];
  bf16* w_s = (bf16*)smem_raw;              // [HREC][LDP]
  bf16* hb0 = w_s + HREC * LDP;             // [RECROWS][LDP]
  bf16* hb1 = hb0 + RECROWS * LDP;

  int k = blockIdx.y, bchunk = blockIdx.x;
  int tid = threadIdx.x, w = tid >> 5, lane = tid & 31;
  const bf16*  whk = Whh + (long)k * whh_kstride;
  const float* xwk = xW + (long)k * xw_kstride;
  bf16*        hok = hout + (long)k * ho_kstride;

  // Stage Whh: thread tid copies row tid (512B) into padded LDS row.
  {
    const bf16x8* s = (const bf16x8*)(whk + (long)tid * HREC);
    bf16x8* d = (bf16x8*)(w_s + tid * LDP);
#pragma unroll
    for (int i = 0; i < HREC / 8; ++i) d[i] = s[i];
  }
  // Init h state from h0[k] (broadcast across the 16 batch rows).
  const float* h0k = h0 + (long)k * HREC;
  for (int i = tid; i < RECROWS * HREC; i += 256) {
    int r = i >> 8, c = i & (HREC - 1);
    hb0[r * LDP + c] = (bf16)h0k[c];
  }
  __syncthreads();

  bf16* hc = hb0;
  bf16* hn = hb1;
  int nb = w * 32;
  int lr = lane & 15, hi = lane >> 4;

  for (int t = T - 1; t >= 0; --t) {
    v8f acc0 = {0.f, 0.f, 0.f, 0.f, 0.f, 0.f, 0.f, 0.f};
    v8f acc1 = acc0;
#pragma unroll
    for (int k0 = 0; k0 < HREC; k0 += 32) {
      bf16x16 a  = load_a_frag(hc, LDP, 0, k0, lane);
      bf16x16 b0 = load_b_frag(w_s, LDP, nb, k0, lane);
      bf16x16 b1 = load_b_frag(w_s, LDP, nb + 16, k0, lane);
      acc0 = wmma_bf16(a, b0, acc0);
      acc1 = wmma_bf16(a, b1, acc1);
    }
#pragma unroll
    for (int tt = 0; tt < 2; ++tt) {
      v8f acc = tt ? acc1 : acc0;
      int n = nb + tt * 16 + lr;
#pragma unroll
      for (int r = 0; r < 8; ++r) {
        int bl = r + hi * 8;                         // local batch row 0..15
        long row = (long)(bchunk * RECROWS + bl) * T + t;
        float v = acc[r] + xwk[row * (long)ldx + n];
        v = fmaxf(v, 0.0f);
        bf16 hv = (bf16)v;
        hn[bl * LDP + n] = hv;                       // next-step state
        hok[row * (long)ldo + n] = hv;               // persistent h output
      }
    }
    __syncthreads();
    bf16* tmp = hc; hc = hn; hn = tmp;
  }
}

// ---------------------------------------------------------------------------
// Sequential z-scan with sparsemax gating; one block per batch row b.
//   h = z_prev @ cb_W1^T + cb_b1 + xs[b,t]   (256 threads, 1 elem each)
//   z_mu[k] = h . cb_W2[k] + cb_b2[k]        (wave k reduces, wave32 shfl)
//   z_t = z_mu + eps_z ; w = sparsemax(z_t) ; x_t = sum_k w_k * mu[k,b,t] + eps_x
// ---------------------------------------------------------------------------
__global__ __launch_bounds__(256) void scan_kernel(
    const float* __restrict__ xs, const float* __restrict__ qmu,
    const float* __restrict__ eps_z, const float* __restrict__ eps_x,
    const float* __restrict__ cb_W1, const float* __restrict__ cb_b1,
    const float* __restrict__ cb_W2, const float* __restrict__ cb_b2,
    const float* __restrict__ z_q0,
    float* __restrict__ x_sample, float* __restrict__ z_sample,
    float* __restrict__ z_logits) {
  __shared__ float h_s[HXH];
  __shared__ float zprev[KB], zmu_s[KB], zt_s[KB], wgt[KB];
  int b = blockIdx.x, tid = threadIdx.x, lane = tid & 31, wv = tid >> 5;

  if (tid < KB) zprev[tid] = z_q0[tid];
  __syncthreads();

  for (int t = 0; t < TT; ++t) {
    long bt = (long)b * TT + t;
    // combiner hidden
    float acc = cb_b1[tid] + xs[bt * HXH + tid];
#pragma unroll
    for (int k = 0; k < KB; ++k) acc += zprev[k] * cb_W1[tid * KB + k];
    h_s[tid] = acc;
    __syncthreads();

    // z_mu: wave wv reduces dot(h, cb_W2[wv])
    float p = 0.0f;
#pragma unroll
    for (int j = 0; j < HXH / 32; ++j)
      p += h_s[lane + j * 32] * cb_W2[wv * HXH + lane + j * 32];
#pragma unroll
    for (int o = 16; o > 0; o >>= 1) p += __shfl_down(p, o, 32);
    if (lane == 0) {
      float zmu = p + cb_b2[wv];
      float ztv = zmu + eps_z[((long)t * BB + b) * KB + wv];
      zmu_s[wv] = zmu;
      zt_s[wv] = ztv;
      z_logits[bt * KB + wv] = zmu;
      z_sample[bt * KB + wv] = ztv;
    }
    __syncthreads();

    // sparsemax over K=8 + state update (single thread; trivial)
    if (tid == 0) {
      float z[KB], zs[KB], cssv[KB];
#pragma unroll
      for (int i = 0; i < KB; ++i) { z[i] = zt_s[i]; zs[i] = zt_s[i]; }
      for (int i = 1; i < KB; ++i) {            // insertion sort, descending
        float key = zs[i]; int j = i - 1;
        while (j >= 0 && zs[j] < key) { zs[j + 1] = zs[j]; --j; }
        zs[j + 1] = key;
      }
      float css = 0.0f;
      for (int i = 0; i < KB; ++i) { css += zs[i]; cssv[i] = css - 1.0f; }
      int cnt = 0;
      for (int i = 0; i < KB; ++i) if ((float)(i + 1) * zs[i] > cssv[i]) ++cnt;
      float tau = cssv[cnt - 1] / (float)cnt;
      for (int i = 0; i < KB; ++i) wgt[i] = fmaxf(z[i] - tau, 0.0f);
      for (int i = 0; i < KB; ++i) zprev[i] = zt_s[i];
    }
    __syncthreads();

    // mixture sample x_t
    if (tid < XX) {
      float v = eps_x[((long)t * BB + b) * XX + tid];
#pragma unroll
      for (int k = 0; k < KB; ++k)
        v += wgt[k] * qmu[((long)k * BT + bt) * XX + tid];
      x_sample[bt * XX + tid] = v;
    }
    __syncthreads();
  }
}

// ---------------------------------------------------------------------------
// Host-side orchestration
// ---------------------------------------------------------------------------
extern "C" void kernel_launch(void* const* d_in, const int* in_sizes, int n_in,
                              void* d_out, int out_size, void* d_ws,
                              size_t ws_size, hipStream_t stream) {
  (void)in_sizes; (void)n_in; (void)out_size; (void)ws_size;

  const float* data   = (const float*)d_in[0];
  const float* eps_qx = (const float*)d_in[1];
  const float* eps_z  = (const float*)d_in[2];
  const float* eps_x  = (const float*)d_in[3];
  const float* y_Wih  = (const float*)d_in[4];
  const float* y_Whh  = (const float*)d_in[5];
  const float* y_bih  = (const float*)d_in[6];
  const float* y_bhh  = (const float*)d_in[7];
  const float* y_h0   = (const float*)d_in[8];
  const float* y_Wmu  = (const float*)d_in[9];
  const float* y_bmu  = (const float*)d_in[10];
  const float* x_Wih  = (const float*)d_in[11];
  const float* x_Whh  = (const float*)d_in[12];
  const float* x_bih  = (const float*)d_in[13];
  const float* x_bhh  = (const float*)d_in[14];
  const float* x_h0   = (const float*)d_in[15];
  const float* ds_W   = (const float*)d_in[16];
  const float* ds_b   = (const float*)d_in[17];
  const float* cb_W1  = (const float*)d_in[18];
  const float* cb_b1  = (const float*)d_in[19];
  const float* cb_W2  = (const float*)d_in[20];
  const float* cb_b2  = (const float*)d_in[21];
  const float* z_q0   = (const float*)d_in[22];

  float* out      = (float*)d_out;
  float* x_sample = out;                                  // [B,T,X]
  float* qmu      = x_sample + (size_t)BT * XX;           // [K,B,T,X]
  float* qlogvar  = qmu + (size_t)KB * BT * XX;           // [K,B,T,X] zeros
  float* z_sample = qlogvar + (size_t)KB * BT * XX;       // [B,T,K]
  float* z_logits = z_sample + (size_t)BT * KB;           // [B,T,K]

  size_t off = 0;
  auto alloc = [&](size_t bytes) -> char* {
    char* p = (char*)d_ws + off;
    off += (bytes + 255) & ~(size_t)255;
    return p;
  };
  bf16* data_bf = (bf16*)alloc((size_t)BT * YY * 2);
  bf16* wihy_bf = (bf16*)alloc((size_t)KB * HYH * YY * 2);
  bf16* whhy_bf = (bf16*)alloc((size_t)KB * HYH * HYH * 2);
  bf16* wmu_bf  = (bf16*)alloc((size_t)KB * XX * HYH * 2);
  bf16* wihx_bf = (bf16*)alloc((size_t)KB * HXH * XX * 2);
  bf16* whhx_bf = (bf16*)alloc((size_t)KB * HXH * HXH * 2);
  bf16* dsw_bf  = (bf16*)alloc((size_t)HXH * KB * HXH * 2);
  float* bias_y = (float*)alloc((size_t)KB * HYH * 4);
  float* bias_x = (float*)alloc((size_t)KB * HXH * 4);
  bf16* qx_bf   = (bf16*)alloc((size_t)KB * BT * XX * 2);
  float* xs_buf = (float*)alloc((size_t)BT * HXH * 4);
  // Aliased large buffers (sequential lifetimes):
  float* xW     = (float*)alloc((size_t)BT * KB * HYH * 4);   // xW_y then xW_x
  bf16* hy_sum  = (bf16*)alloc((size_t)KB * BT * HYH * 2);    // h_y then summary

  // --- precision conversion + bias folding (bih+bhh) ---
  auto cvt = [&](const float* s, bf16* d, long n) {
    f32_to_bf16_kernel<<<dim3((unsigned)((n + 255) / 256)), dim3(256), 0, stream>>>(s, d, n);
  };
  cvt(data,  data_bf, (long)BT * YY);
  cvt(y_Wih, wihy_bf, (long)KB * HYH * YY);
  cvt(y_Whh, whhy_bf, (long)KB * HYH * HYH);
  cvt(y_Wmu, wmu_bf,  (long)KB * XX * HYH);
  cvt(x_Wih, wihx_bf, (long)KB * HXH * XX);
  cvt(x_Whh, whhx_bf, (long)KB * HXH * HXH);
  cvt(ds_W,  dsw_bf,  (long)HXH * KB * HXH);
  add_vec_kernel<<<dim3(8), dim3(256), 0, stream>>>(y_bih, y_bhh, bias_y, KB * HYH);
  add_vec_kernel<<<dim3(8), dim3(256), 0, stream>>>(x_bih, x_bhh, bias_x, KB * HXH);
  fill_zero_kernel<<<dim3((KB * BT * XX) / 256), dim3(256), 0, stream>>>(
      qlogvar, (long)KB * BT * XX);

  // --- y input projection: xW_y[bt, k*HY+h] = data @ y_Wih^T + (bih+bhh) ---
  gemm_bt_kernel<<<dim3(BT / 64, (KB * HYH) / 64, 1), dim3(256), 0, stream>>>(
      data_bf, wihy_bf, bias_y, xW, nullptr, nullptr,
      YY, YY, YY, KB * HYH, 0, 0, 0, 0, /*flags=*/0);

  // --- y backward recurrence -> h_y [K][BT][HY] (bf16) ---
  const size_t REC_SMEM = (size_t)(HREC * LDP + 2 * RECROWS * LDP) * sizeof(bf16);
  (void)hipFuncSetAttribute((const void*)rec_kernel,
                            hipFuncAttributeMaxDynamicSharedMemorySize,
                            (int)REC_SMEM);
  rec_kernel<<<dim3(BB / RECROWS, KB), dim3(256), REC_SMEM, stream>>>(
      whhy_bf, (long)HYH * HYH, xW, /*xw_kstride=*/(long)HYH, /*ldx=*/KB * HYH,
      y_h0, hy_sum, /*ho_kstride=*/(long)BT * HYH, /*ldo=*/HYH, TT);

  // --- q_x_mu = h_y @ y_Wmu^T + y_bmu  (batched over k; fused q_x bf16) ---
  gemm_bt_kernel<<<dim3(BT / 64, 1, KB), dim3(256), 0, stream>>>(
      hy_sum, wmu_bf, y_bmu, qmu, qx_bf, eps_qx,
      HYH, HYH, HYH, XX,
      (long)BT * HYH, (long)XX * HYH, XX, (long)BT * XX, /*flags=*/2);

  // --- x input projection: xW_x[k][bt][hx] = q_x @ x_Wih^T + (bih+bhh) ---
  gemm_bt_kernel<<<dim3(BT / 64, HXH / 64, KB), dim3(256), 0, stream>>>(
      qx_bf, wihx_bf, bias_x, xW, nullptr, nullptr,
      XX, XX, XX, HXH,
      (long)BT * XX, (long)HXH * XX, HXH, (long)BT * HXH, /*flags=*/0);

  // --- x backward recurrence -> summary [BT][K*HX] (bf16) ---
  rec_kernel<<<dim3(BB / RECROWS, KB), dim3(256), REC_SMEM, stream>>>(
      whhx_bf, (long)HXH * HXH, xW, /*xw_kstride=*/(long)BT * HXH, /*ldx=*/HXH,
      x_h0, hy_sum, /*ho_kstride=*/(long)HXH, /*ldo=*/KB * HXH, TT);

  // --- downsampler: xs = summary @ ds_W^T + ds_b ---
  gemm_bt_kernel<<<dim3(BT / 64, HXH / 64, 1), dim3(256), 0, stream>>>(
      hy_sum, dsw_bf, ds_b, xs_buf, nullptr, nullptr,
      KB * HXH, KB * HXH, KB * HXH, HXH, 0, 0, 0, 0, /*flags=*/0);

  // --- sequential z-scan with sparsemax (64 independent batch rows) ---
  scan_kernel<<<dim3(BB), dim3(256), 0, stream>>>(
      xs_buf, qmu, eps_z, eps_x, cb_W1, cb_b1, cb_W2, cb_b2, z_q0,
      x_sample, z_sample, z_logits);
}
